// NpuWeightQuantBatchmatmulModule_32023276159083
// MI455X (gfx1250) — matrix operand order repaired and checked by reading, not compiled, not run on hardware
//
#include <hip/hip_runtime.h>

typedef __attribute__((ext_vector_type(16))) __bf16 v16bf;
typedef __attribute__((ext_vector_type(8)))  float  v8f;
typedef __attribute__((ext_vector_type(4)))  unsigned int u32x4;
typedef __attribute__((ext_vector_type(8)))  int i32x8;
typedef __attribute__((ext_vector_type(4)))  int i32x4;

#define MDIM 2048
#define KDIM 4096
#define NDIM 11008

#define BM 128
#define BN 128
#define BK 32
#define ASTR 40                 // A row stride in bf16 (32 + 8 pad): 80 B rows, conflict-free frag loads
#define BSTR 36                 // B col stride in bf16 (32 + 4 pad): 8B aligned, conflict-free loads
#define ATILE (BM * ASTR)       // 5120 elems
#define BTILE (BN * BSTR)       // 4608 elems
#define BUFE  (ATILE + BTILE)   // 9728 elems per buffer

__device__ __forceinline__ float bf2f(unsigned short u) {
    unsigned int v = ((unsigned int)u) << 16;
    return __builtin_bit_cast(float, v);
}
__device__ __forceinline__ unsigned short f2bf(float f) {
    __bf16 b = (__bf16)f;
    return __builtin_bit_cast(unsigned short, b);
}

// ---- Tensor Data Mover: DMA one 128-row x 64-byte A tile into LDS.
// 2D descriptor, data_size = 8B. LDS pad: after every 16 DWORDs (64B row)
// insert 4 DWORDs (16B) -> effective LDS row stride 80 B == ASTR*2.
__device__ __forceinline__ void tdm_load_A(const unsigned short* gsrc, unsigned int lds_byte_off) {
    unsigned long long ga = (unsigned long long)(size_t)gsrc;

    u32x4 g0;
    g0.x = 1u;                                               // count=1, user descriptor
    g0.y = lds_byte_off;                                     // lds_addr [63:32]
    g0.z = (unsigned int)ga;                                 // global_addr [95:64]
    g0.w = ((unsigned int)(ga >> 32) & 0x01ffffffu)          // global_addr [120:96]
         | (2u << 30);                                       // type=2 (image)

    i32x8 g1;
    g1[0] = (3 << 16)      // data_size = 8 bytes
          | (1 << 20)      // pad_enable
          | (3 << 22)      // pad_interval: 16 DWORDs
          | (3 << 25);     // pad_amount:   4 DWORDs
    g1[1] = (int)((((unsigned)(KDIM * 2 / 8)) & 0xffffu) << 16);  // tensor_dim0 lo16 (1024 qwords)
    g1[2] = (int)((((unsigned)MDIM) & 0xffffu) << 16);            // dim0 hi16=0 | tensor_dim1 lo16
    g1[3] = (8 << 16);                                            // dim1 hi16=0 | tile_dim0 = 8 qwords (64B)
    g1[4] = BM;                                                   // tile_dim1 = 128 rows | tile_dim2 = 0
    g1[5] = (int)(KDIM * 2 / 8);                                  // tensor_dim0_stride = 1024 qwords
    g1[6] = 0;
    g1[7] = 0;

    i32x4 gz = {0, 0, 0, 0};
#if defined(__clang_major__) && (__clang_major__ >= 23)
    i32x8 gz8 = {0, 0, 0, 0, 0, 0, 0, 0};
    __builtin_amdgcn_tensor_load_to_lds(g0, g1, gz, gz, gz8, 0);
#else
    __builtin_amdgcn_tensor_load_to_lds(g0, g1, gz, gz, 0);
#endif
}

__global__ __launch_bounds__(256)
void wq_bmm_kernel(const unsigned short* __restrict__ x,
                   const signed char*    __restrict__ w,
                   const unsigned short* __restrict__ scale,
                   const unsigned short* __restrict__ offs,
                   unsigned short*       __restrict__ out)
{
    __shared__ unsigned short lds[2 * BUFE];   // ~38 KB, double buffered

    const int tid  = threadIdx.x;
    const int lane = tid & 31;
    const int wid  = tid >> 5;
    const int wm   = wid & 3;    // wave row group: 32 rows (2 M-tiles)
    const int wn   = wid >> 2;   // wave col group: 64 cols (4 N-tiles)

    const int bm = blockIdx.y * BM;
    const int bn = blockIdx.x * BN;

    const unsigned int lds_base = (unsigned int)(size_t)(void*)&lds[0];

    // ---- B staging: thread handles 4 k-rows x 4 cols micro-tile
    const int bk0 = (tid >> 5) << 2;      // 0..28 step 4
    const int bc0 = (tid & 31) << 2;      // 0..124 step 4

    // per-channel dequant constants: (q + off)*s == ubyte*s + (off-128)*s
    float sc[4], cb[4];
#pragma unroll
    for (int j = 0; j < 4; ++j) {
        sc[j] = bf2f(scale[bn + bc0 + j]);
        cb[j] = (bf2f(offs[bn + bc0 + j]) - 128.0f) * sc[j];
    }

    unsigned int wq[4];

    auto gloadB = [&](int kt) {
#pragma unroll
        for (int i = 0; i < 4; ++i)
            wq[i] = *(const unsigned int*)(w + (size_t)(kt * BK + bk0 + i) * NDIM + bn + bc0)
                    ^ 0x80808080u;   // int8 -> biased uint8 (enables v_cvt_f32_ubyteN)
    };

    auto lstoreB = [&](int buf) {
        unsigned short* B = lds + buf * BUFE + ATILE;
#pragma unroll
        for (int j = 0; j < 4; ++j) {
            unsigned long long pk = 0ull;
#pragma unroll
            for (int i = 0; i < 4; ++i) {
                float f = (float)((wq[i] >> (8 * j)) & 0xffu);
                float d = __builtin_fmaf(f, sc[j], cb[j]);
                pk |= ((unsigned long long)f2bf(d)) << (16 * i);
            }
            *(unsigned long long*)(B + (bc0 + j) * BSTR + bk0) = pk;
        }
    };

    v8f acc[8] = {};   // 2 (M) x 4 (N) tiles of 16x16 f32

    union AF { uint4 q[2];              v16bf v; };
    union BF { unsigned long long d[4]; v16bf v; };

    auto compute = [&](int buf) {
        const unsigned short* A = lds + buf * BUFE;
        const unsigned short* B = A + ATILE;
        const int l16 = lane & 15;

        // A fragments: lane<16 holds K {0..7,16..23}; lane>=16 holds K {8..15,24..31}
        AF af[2];
        const int ak0 = (lane < 16) ? 0 : 8;
#pragma unroll
        for (int tm = 0; tm < 2; ++tm) {
            const unsigned short* p = A + (wm * 32 + tm * 16 + l16) * ASTR + ak0;
            af[tm].q[0] = *(const uint4*)(p);
            af[tm].q[1] = *(const uint4*)(p + 16);
        }

        // B fragments: lane = column; lane<16 -> K {0..7,16..23}, lane>=16 -> K {8..15,24..31}
        BF bf[4];
        const int bko = (lane < 16) ? 0 : 8;
#pragma unroll
        for (int tn = 0; tn < 4; ++tn) {
            const unsigned short* p = B + (wn * 64 + tn * 16 + l16) * BSTR + bko;
            bf[tn].d[0] = *(const unsigned long long*)(p);
            bf[tn].d[1] = *(const unsigned long long*)(p + 4);
            bf[tn].d[2] = *(const unsigned long long*)(p + 16);
            bf[tn].d[3] = *(const unsigned long long*)(p + 20);
        }

#pragma unroll
        for (int tm = 0; tm < 2; ++tm)
#pragma unroll
            for (int tn = 0; tn < 4; ++tn) {
                const int t = tm * 4 + tn;
                acc[t] = __builtin_amdgcn_wmma_f32_16x16x32_bf16(
                    false, af[tm].v, false, bf[tn].v, (short)0, acc[t], false, false);
            }
    };

    // ---- software-pipelined main loop: TDM moves A, threads dequant-stage B ----
    const unsigned short* aTile = x + (size_t)bm * KDIM;

    if (wid == 0) tdm_load_A(aTile, lds_base);
    gloadB(0);
    lstoreB(0);
    if (wid == 0) __builtin_amdgcn_s_wait_tensorcnt(0);
    __syncthreads();

    const int NKT = KDIM / BK;   // 128
    for (int kt = 0; kt < NKT; ++kt) {
        const bool more = (kt + 1 < NKT);
        const int nxt = (kt + 1) & 1;
        if (more) {
            if (wid == 0)
                tdm_load_A(aTile + (kt + 1) * BK, lds_base + nxt * (BUFE * 2));
            gloadB(kt + 1);                 // global prefetch overlaps compute
        }
        compute(kt & 1);
        if (more) lstoreB(nxt);             // fill other buffer
        if (more && wid == 0) __builtin_amdgcn_s_wait_tensorcnt(0);
        __syncthreads();
    }

    // ---- epilogue: f32 -> bf16; C layout: VGPR j -> row j (+8 for lanes>=16), col = lane&15
#pragma unroll
    for (int tm = 0; tm < 2; ++tm) {
#pragma unroll
        for (int tn = 0; tn < 4; ++tn) {
            const int t = tm * 4 + tn;
            const int row0 = bm + wm * 32 + tm * 16 + ((lane >> 4) << 3);
            const int col  = bn + wn * 64 + tn * 16 + (lane & 15);
#pragma unroll
            for (int j = 0; j < 8; ++j)
                out[(size_t)(row0 + j) * NDIM + col] = f2bf(acc[t][j]);
        }
    }
}

extern "C" void kernel_launch(void* const* d_in, const int* in_sizes, int n_in,
                              void* d_out, int out_size, void* d_ws, size_t ws_size,
                              hipStream_t stream) {
    const unsigned short* x = (const unsigned short*)d_in[0];
    const signed char*    w = (const signed char*)d_in[1];
    const unsigned short* s = (const unsigned short*)d_in[2];
    const unsigned short* o = (const unsigned short*)d_in[3];
    unsigned short* outp = (unsigned short*)d_out;

    dim3 grid(NDIM / BN, MDIM / BM);   // (86, 16) blocks
    wq_bmm_kernel<<<grid, dim3(256), 0, stream>>>(x, w, s, o, outp);
}
